// SimpleFrequencyAttention_21981642621476
// MI455X (gfx1250) — matrix operand order repaired and checked by reading, not compiled, not run on hardware
//
#include <hip/hip_runtime.h>

// ---------------------------------------------------------------------------
// SimpleFrequencyAttention for MI455X (gfx1250, wave32, WMMA bf16)
//   x[8,128,64,64] f32, W_qkv[384,128], W_out[128,128], b_out[128]
//   Flash-attention fused pipeline, bf16 WMMA (16x16x32) with f32 accum.
// ---------------------------------------------------------------------------

typedef __attribute__((ext_vector_type(16))) __bf16 bf16x16;
typedef __attribute__((ext_vector_type(2)))  __bf16 bf16x2;
typedef __attribute__((ext_vector_type(8)))  float  floatx8;

union Frag {                 // 16 bf16 = 32 bytes = 8 VGPRs
    unsigned u[8];
    uint4    q[2];
};

static __device__ __forceinline__ bf16x16 as_bf(const Frag& x) {
    return __builtin_bit_cast(bf16x16, x);
}

static __device__ __forceinline__ unsigned short f32_to_bf16(float f) {
    return __builtin_bit_cast(unsigned short, (__bf16)f);   // hw cvt, RNE
}

static __device__ __forceinline__ unsigned pack_bf16(float a, float b) {
    bf16x2 p;
    p.x = (__bf16)a;
    p.y = (__bf16)b;
    return __builtin_bit_cast(unsigned, p);                 // v_cvt_pk_bf16_f32
}

static __device__ __forceinline__ floatx8 wmma_bf(const Frag& A, const Frag& B, floatx8 C) {
    // D = A(16x32 bf16) * B(32x16 bf16) + C(16x16 f32)
    return __builtin_amdgcn_wmma_f32_16x16x32_bf16(false, as_bf(A), false, as_bf(B),
                                                   (short)0, C, false, false);
}

#define BATCH 8
#define CCH   128
#define NPIX  4096
#define BSTRT ((size_t)NPIX * CCH)   // per-batch elements of [n,c] / [c,n] buffers

// ---------------------------------------------------------------------------
// Kernel 1: QKV projection. qkv[o,i] = sum_c Wqkv[o,c] * x[b,c,i]
//   o in [0,128)   -> Qt[b][i][c]  (transposed, bf16)
//   o in [128,256) -> Kt[b][i][c]  (transposed, bf16)
//   o in [256,384) -> Vk[b][c][i]  (natural,    bf16)
// ---------------------------------------------------------------------------
__global__ __launch_bounds__(128) void qkv_proj_kernel(
    const float* __restrict__ x, const float* __restrict__ Wqkv,
    unsigned short* __restrict__ Qt, unsigned short* __restrict__ Kt,
    unsigned short* __restrict__ Vk)
{
    const int lane = threadIdx.x & 31;
    const int wave = threadIdx.x >> 5;
    const int wg   = blockIdx.x * 4 + wave;
    const int lo = lane & 15, hi = lane >> 4;

    const int ot  = wg % 24;           // 384/16 o-tiles
    const int rem = wg / 24;
    const int it  = rem % 256;         // 4096/16 i-tiles
    const int b   = rem / 256;
    const int o0 = ot * 16, i0 = it * 16;

    Frag a[4], bx[4];
    #pragma unroll
    for (int s = 0; s < 4; ++s) {
        // A fragment from fp32 weights: lane's 8 dwords are 2 runs of 8 floats
        const float* wrow = Wqkv + (size_t)(o0 + lo) * CCH + s * 32 + hi * 8;
        const float4 w0 = *(const float4*)(wrow);
        const float4 w1 = *(const float4*)(wrow + 4);
        const float4 w2 = *(const float4*)(wrow + 16);
        const float4 w3 = *(const float4*)(wrow + 20);
        a[s].u[0] = pack_bf16(w0.x, w0.y);  a[s].u[1] = pack_bf16(w0.z, w0.w);
        a[s].u[2] = pack_bf16(w1.x, w1.y);  a[s].u[3] = pack_bf16(w1.z, w1.w);
        a[s].u[4] = pack_bf16(w2.x, w2.y);  a[s].u[5] = pack_bf16(w2.z, w2.w);
        a[s].u[6] = pack_bf16(w3.x, w3.y);  a[s].u[7] = pack_bf16(w3.z, w3.w);
        // B fragment from x[b,c,i]: k (=c) runs along rows -> strided loads
        #pragma unroll
        for (int v = 0; v < 8; ++v) {
            const int kb = s * 32 + hi * 8 + (v & 3) * 2 + (v >> 2) * 16;
            const float* xp = x + ((size_t)b * CCH + kb) * NPIX + (i0 + lo);
            bx[s].u[v] = pack_bf16(xp[0], xp[NPIX]);
        }
    }

    floatx8 acc = {0.f, 0.f, 0.f, 0.f, 0.f, 0.f, 0.f, 0.f};
    #pragma unroll
    for (int s = 0; s < 4; ++s) acc = wmma_bf(a[s], bx[s], acc);

    #pragma unroll
    for (int v = 0; v < 8; ++v) {
        const int m = v + hi * 8;       // C/D: m = v + 8*(lane/16)
        const int o = o0 + m;
        const int i = i0 + lo;
        const unsigned short h = f32_to_bf16(acc[v]);
        if (ot < 8)
            Qt[(size_t)b * BSTRT + (size_t)i * CCH + o] = h;
        else if (ot < 16)
            Kt[(size_t)b * BSTRT + (size_t)i * CCH + (o - 128)] = h;
        else
            Vk[(size_t)b * BSTRT + (size_t)(o - 256) * NPIX + i] = h;
    }
}

// ---------------------------------------------------------------------------
// Kernel 2: flash attention. One wave owns 32 queries (2 x 16-row M-tiles),
// streams key tiles of 32 with online softmax; O accumulated in f32 regs.
// All bf16 fragments load as 2 x b128 per lane (contiguous, 16B aligned).
// ---------------------------------------------------------------------------
__global__ __launch_bounds__(128, 1) void flash_attn_kernel(
    const unsigned short* __restrict__ Qt, const unsigned short* __restrict__ Kt,
    const unsigned short* __restrict__ Vk, unsigned short* __restrict__ Ot)
{
    __shared__ __align__(16) unsigned short p_lds[4][16 * 32];  // per-wave P tile

    const int lane = threadIdx.x & 31;
    const int wave = threadIdx.x >> 5;
    const int wg   = blockIdx.x * 4 + wave;
    const int lo = lane & 15, hi = lane >> 4;

    const int qg = wg & 127;            // 4096/32 query groups per batch
    const int b  = wg >> 7;
    const int i0 = qg * 32;

    const unsigned short* qtb = Qt + (size_t)b * BSTRT;
    const unsigned short* ktb = Kt + (size_t)b * BSTRT;
    const unsigned short* vkb = Vk + (size_t)b * BSTRT;
    unsigned short* pl = p_lds[wave];
    const float scale2 = 0.08838834764831845f * 1.4426950408889634f;  // c^-0.5 * log2(e)

    // Preload Q A-fragments: 2 M-tiles x 4 K-steps (c = 0..127)
    Frag qa[2][4];
    #pragma unroll
    for (int t = 0; t < 2; ++t)
        #pragma unroll
        for (int s = 0; s < 4; ++s) {
            const unsigned short* qp = qtb + (size_t)(i0 + t * 16 + lo) * CCH + s * 32 + hi * 8;
            qa[t][s].q[0] = *(const uint4*)(qp);
            qa[t][s].q[1] = *(const uint4*)(qp + 16);
        }

    float rm[2][8], rs[2][8];
    floatx8 oacc[2][8];
    #pragma unroll
    for (int t = 0; t < 2; ++t)
        #pragma unroll
        for (int v = 0; v < 8; ++v) {
            rm[t][v] = -1e30f;
            rs[t][v] = 0.f;
        }
    #pragma unroll
    for (int t = 0; t < 2; ++t)
        #pragma unroll
        for (int ct = 0; ct < 8; ++ct)
            oacc[t][ct] = (floatx8){0.f, 0.f, 0.f, 0.f, 0.f, 0.f, 0.f, 0.f};

    for (int j0 = 0; j0 < NPIX; j0 += 32) {
        // K B-fragments (S = Q^T K): 4 c-steps x 2 key halves
        Frag kb[4][2], vb[8];
        #pragma unroll
        for (int s = 0; s < 4; ++s)
            #pragma unroll
            for (int h = 0; h < 2; ++h) {
                const unsigned short* kp =
                    ktb + (size_t)(j0 + h * 16 + lo) * CCH + s * 32 + hi * 8;
                kb[s][h].q[0] = *(const uint4*)(kp);
                kb[s][h].q[1] = *(const uint4*)(kp + 16);
            }
        // V B-fragments (O += P V^T): 8 c-tiles, k = key index (table order)
        #pragma unroll
        for (int ct = 0; ct < 8; ++ct) {
            const unsigned short* vp =
                vkb + (size_t)(ct * 16 + lo) * NPIX + j0 + hi * 8;
            vb[ct].q[0] = *(const uint4*)(vp);
            vb[ct].q[1] = *(const uint4*)(vp + 16);
        }

        #pragma unroll
        for (int t = 0; t < 2; ++t) {
            floatx8 s0 = {0.f, 0.f, 0.f, 0.f, 0.f, 0.f, 0.f, 0.f};
            floatx8 s1 = {0.f, 0.f, 0.f, 0.f, 0.f, 0.f, 0.f, 0.f};
            #pragma unroll
            for (int s = 0; s < 4; ++s) {
                s0 = wmma_bf(qa[t][s], kb[s][0], s0);
                s1 = wmma_bf(qa[t][s], kb[s][1], s1);
            }

            // Online softmax in exp2 domain: row m = v + 8*hi across 16-lane half
            float p0[8], p1[8], alpha[8];
            #pragma unroll
            for (int v = 0; v < 8; ++v) {
                const float a0 = s0[v] * scale2, a1 = s1[v] * scale2;
                float mx = fmaxf(a0, a1);
                mx = fmaxf(mx, __shfl_xor(mx, 1));
                mx = fmaxf(mx, __shfl_xor(mx, 2));
                mx = fmaxf(mx, __shfl_xor(mx, 4));
                mx = fmaxf(mx, __shfl_xor(mx, 8));
                const float mnew = fmaxf(rm[t][v], mx);
                alpha[v] = __builtin_amdgcn_exp2f(rm[t][v] - mnew);
                rm[t][v] = mnew;
                p0[v] = __builtin_amdgcn_exp2f(a0 - mnew);
                p1[v] = __builtin_amdgcn_exp2f(a1 - mnew);
                float sm = p0[v] + p1[v];
                sm += __shfl_xor(sm, 1);
                sm += __shfl_xor(sm, 2);
                sm += __shfl_xor(sm, 4);
                sm += __shfl_xor(sm, 8);
                rs[t][v] = rs[t][v] * alpha[v] + sm;
            }
            #pragma unroll
            for (int ct = 0; ct < 8; ++ct)
                #pragma unroll
                for (int v = 0; v < 8; ++v)
                    oacc[t][ct][v] *= alpha[v];

            // Bounce P through LDS: write in C/D layout, read back in A layout
            #pragma unroll
            for (int v = 0; v < 8; ++v) {
                const int m = v + hi * 8;
                pl[m * 32 + lo]      = f32_to_bf16(p0[v]);
                pl[m * 32 + 16 + lo] = f32_to_bf16(p1[v]);
            }
            __builtin_amdgcn_wave_barrier();
            Frag pa;
            pa.q[0] = *(const uint4*)(pl + lo * 32 + hi * 8);
            pa.q[1] = *(const uint4*)(pl + lo * 32 + 16 + hi * 8);
            __builtin_amdgcn_wave_barrier();

            #pragma unroll
            for (int ct = 0; ct < 8; ++ct)
                oacc[t][ct] = wmma_bf(pa, vb[ct], oacc[t][ct]);
        }
    }

    // Normalize (rcp once per row) and store O^T as bf16 [n, c]
    float inv[2][8];
    #pragma unroll
    for (int t = 0; t < 2; ++t)
        #pragma unroll
        for (int v = 0; v < 8; ++v)
            inv[t][v] = __builtin_amdgcn_rcpf(rs[t][v]);

    const size_t obase = (size_t)b * BSTRT;
    #pragma unroll
    for (int t = 0; t < 2; ++t)
        #pragma unroll
        for (int ct = 0; ct < 8; ++ct)
            #pragma unroll
            for (int v = 0; v < 8; ++v) {
                const int i = i0 + t * 16 + v + hi * 8;
                const int c = ct * 16 + lo;
                Ot[obase + (size_t)i * CCH + c] = f32_to_bf16(oacc[t][ct][v] * inv[t][v]);
            }
}

// ---------------------------------------------------------------------------
// Kernel 3: output projection. y[b,o,i] = sum_c Wout[o,c] * O[b,c,i] + b_out[o]
// ---------------------------------------------------------------------------
__global__ __launch_bounds__(128) void out_proj_kernel(
    const unsigned short* __restrict__ Ot, const float* __restrict__ Wout,
    const float* __restrict__ bout, float* __restrict__ y)
{
    const int lane = threadIdx.x & 31;
    const int wave = threadIdx.x >> 5;
    const int wg   = blockIdx.x * 4 + wave;
    const int lo = lane & 15, hi = lane >> 4;

    const int ot = wg & 7;              // 128/16 o-tiles
    const int it = (wg >> 3) & 255;     // 4096/16 i-tiles
    const int b  = wg >> 11;
    const int o0 = ot * 16, i0 = it * 16;

    Frag a[4], bo[4];
    #pragma unroll
    for (int s = 0; s < 4; ++s) {
        const float* wrow = Wout + (size_t)(o0 + lo) * CCH + s * 32 + hi * 8;
        const float4 w0 = *(const float4*)(wrow);
        const float4 w1 = *(const float4*)(wrow + 4);
        const float4 w2 = *(const float4*)(wrow + 16);
        const float4 w3 = *(const float4*)(wrow + 20);
        a[s].u[0] = pack_bf16(w0.x, w0.y);  a[s].u[1] = pack_bf16(w0.z, w0.w);
        a[s].u[2] = pack_bf16(w1.x, w1.y);  a[s].u[3] = pack_bf16(w1.z, w1.w);
        a[s].u[4] = pack_bf16(w2.x, w2.y);  a[s].u[5] = pack_bf16(w2.z, w2.w);
        a[s].u[6] = pack_bf16(w3.x, w3.y);  a[s].u[7] = pack_bf16(w3.z, w3.w);

        const unsigned short* op =
            Ot + (size_t)b * BSTRT + (size_t)(i0 + lo) * CCH + s * 32 + hi * 8;
        bo[s].q[0] = *(const uint4*)(op);
        bo[s].q[1] = *(const uint4*)(op + 16);
    }

    floatx8 acc = {0.f, 0.f, 0.f, 0.f, 0.f, 0.f, 0.f, 0.f};
    #pragma unroll
    for (int s = 0; s < 4; ++s) acc = wmma_bf(a[s], bo[s], acc);

    #pragma unroll
    for (int v = 0; v < 8; ++v) {
        const int o = o0 + v + hi * 8;
        y[((size_t)b * CCH + o) * NPIX + i0 + lo] = acc[v] + bout[o];
    }
}

// ---------------------------------------------------------------------------
extern "C" void kernel_launch(void* const* d_in, const int* in_sizes, int n_in,
                              void* d_out, int out_size, void* d_ws, size_t ws_size,
                              hipStream_t stream) {
    const float* x    = (const float*)d_in[0];   // [8,128,64,64]
    const float* Wqkv = (const float*)d_in[1];   // [384,128]
    const float* Wout = (const float*)d_in[2];   // [128,128]
    const float* bout = (const float*)d_in[3];   // [128]
    float* y = (float*)d_out;                    // [8,128,64,64]

    unsigned short* ws = (unsigned short*)d_ws;
    const size_t per = (size_t)BATCH * BSTRT;    // 4,194,304 bf16 elements each
    unsigned short* Qt = ws;                     // [b][n][c]
    unsigned short* Kt = ws + per;               // [b][n][c]
    unsigned short* Vk = ws + 2 * per;           // [b][c][n]
    unsigned short* Ot = ws + 3 * per;           // [b][n][c]

    // 1) QKV projection: 24*256*8 = 49152 wave-tiles, 4 waves/block
    qkv_proj_kernel<<<49152 / 4, 128, 0, stream>>>(x, Wqkv, Qt, Kt, Vk);
    // 2) Flash attention: 8 batches * 128 query-groups = 1024 waves
    flash_attn_kernel<<<1024 / 4, 128, 0, stream>>>(Qt, Kt, Vk, Ot);
    // 3) Output projection: 8*256*8 = 16384 wave-tiles
    out_proj_kernel<<<16384 / 4, 128, 0, stream>>>(Ot, Wout, bout, y);
}
